// EdgeSelfAttention_15539191677812
// MI455X (gfx1250) — compile-verified, matrix operand order checked by reading, not run on hardware
//
#include <hip/hip_runtime.h>
#include <hip/hip_bf16.h>

typedef __bf16 bf16;
typedef __attribute__((ext_vector_type(16))) __bf16 v16bf;
typedef __attribute__((ext_vector_type(8)))  float  v8f;
typedef __attribute__((ext_vector_type(4)))  unsigned int uint4v;

union F16frag {
    v16bf  v;
    uint4v u[2];
};

__device__ __forceinline__ bf16 f2bf(float f) {
    unsigned u = __builtin_bit_cast(unsigned, f);
    u = (u + 0x7FFFu + ((u >> 16) & 1u)) >> 16;
    return __builtin_bit_cast(bf16, (unsigned short)u);
}

__device__ __forceinline__ v8f wmma_bf16(v16bf a, v16bf b, v8f c) {
    // D = A(16x32 bf16) * B(32x16 bf16) + C(16x16 f32)
    return __builtin_amdgcn_wmma_f32_16x16x32_bf16(
        false, a, false, b, (short)0, c, false, false);
}

// Fragment loader (A row-major / B^T row-major share the same VGPR layout):
// lanes take K = {k0+half*8 .. +7, k0+16+half*8 .. +7} from row `rowptr`.
__device__ __forceinline__ v16bf load_frag(const bf16* rowptr, int k0, int half) {
    F16frag f;
    f.u[0] = *(const uint4v*)(rowptr + k0 + half * 8);
    f.u[1] = *(const uint4v*)(rowptr + k0 + 16 + half * 8);
    return f.v;
}

// ---------------- weight packing: fp32 OIHW -> bf16, K' = tap*C + c ----------
__global__ void pack_w1_kernel(const float* __restrict__ w, bf16* __restrict__ out) {
    int t = blockIdx.x * 256 + threadIdx.x;
    if (t >= 128 * 160) return;
    int m = t / 160, k = t % 160;
    float v = 0.f;
    if (k < 144) {
        int p = k / 16, c = k % 16;              // p = kh*3+kw
        v = w[m * 144 + c * 9 + p];
    }
    out[t] = f2bf(v);
}

__global__ void pack_w2_kernel(const float* __restrict__ w, bf16* __restrict__ out) {
    int t = blockIdx.x * 256 + threadIdx.x;
    if (t >= 128 * 1152) return;
    int m = t / 1152, k = t % 1152;
    int p = k / 128, c = k % 128;
    out[t] = f2bf(w[m * 1152 + c * 9 + p]);
}

// ---------------- fused conv block: one block per edge ----------------------
__global__ __launch_bounds__(256) void conv_block_kernel(
    const float* __restrict__ ef,   // [2048][16][256]
    const bf16*  __restrict__ w1r,  // [128][160]
    const float* __restrict__ b1,   // [128]
    const bf16*  __restrict__ w2r,  // [128][1152]
    const float* __restrict__ b2,   // [128]
    bf16* __restrict__ outb)        // [2048][18432]
{
    __shared__ __align__(16) bf16 inT[256][16];    // [spatial][cin]     8 KB
    __shared__ __align__(16) bf16 H1t[196][128];   // [sp 14x14][cout]  50 KB
    __shared__ __align__(16) bf16 H2[128][144];    // [cout][sp 12x12]  36 KB

    const int e    = blockIdx.x;
    const int lane = threadIdx.x & 31;
    const int wid  = threadIdx.x >> 5;   // 8 waves
    const int half = lane >> 4;
    const int r    = lane & 15;
    const int m0   = wid * 16;           // cout strip

    // stage input, fp32 -> bf16, transposed
    for (int i = threadIdx.x; i < 4096; i += 256) {
        int c = i >> 8, s = i & 255;
        inT[s][c] = f2bf(ef[(size_t)e * 4096 + i]);
    }
    __syncthreads();

    const uint4v z4 = {0u, 0u, 0u, 0u};

    float bias1[8], bias2[8];
    #pragma unroll
    for (int i = 0; i < 8; ++i) {
        bias1[i] = b1[m0 + half * 8 + i];
        bias2[i] = b2[m0 + half * 8 + i];
    }

    // ---------------- conv1: M=128, K=160(pad), N=196 (13 tiles) ------------
    for (int nt = 0; nt < 13; ++nt) {
        v8f acc = {};
        const int n   = nt * 16 + r;                  // output spatial (14x14)
        const int n16 = (n / 14) * 16 + (n % 14);     // input base index
        const bool ok = (n < 196);
        // per-lane kt-invariant B pointer; per-tap offset is uniform
        const bf16* bbase = &inT[0][0] + n16 * 16 + half * 8;
        #pragma unroll
        for (int kt = 0; kt < 5; ++kt) {              // K chunks of 32
            F16frag a, b;
            const bf16* ap = w1r + (m0 + r) * 160 + kt * 32;
            a.u[0] = *(const uint4v*)(ap + half * 8);
            a.u[1] = *(const uint4v*)(ap + 16 + half * 8);
            #pragma unroll
            for (int g2 = 0; g2 < 2; ++g2) {          // two taps per k-chunk
                const int p = kt * 2 + g2;            // tap index kh*3+kw
                if (p < 9 && ok) {
                    const int tap16 = (p / 3) * 16 + (p % 3);   // constant
                    b.u[g2] = *(const uint4v*)(bbase + tap16 * 16);
                } else {
                    b.u[g2] = z4;
                }
            }
            acc = wmma_bf16(a.v, b.v, acc);
        }
        if (ok) {
            #pragma unroll
            for (int i = 0; i < 8; ++i) {
                float v = acc[i] + bias1[i];
                H1t[n][m0 + half * 8 + i] = f2bf(v > 0.f ? v : 0.f);
            }
        }
    }
    __syncthreads();

    // ---------------- conv2: M=128, K=1152, N=144 (9 tiles) -----------------
    // kt-outer / nt-inner: A fragment loaded once, used 9 times.
    // B address = H1t + offs[nt] (kt-invariant) + tapoff (lane-invariant).
    int offs[9];
    #pragma unroll
    for (int nt = 0; nt < 9; ++nt) {
        int n = nt * 16 + r;
        offs[nt] = ((n / 12) * 14 + (n % 12)) * 128 + half * 8;
    }
    const bf16* hbase = &H1t[0][0];
    v8f acc2[9] = {};
    for (int kt = 0; kt < 36; ++kt) {                 // K chunks of 32
        const int p  = kt >> 2;                       // tap (4 chunks per tap)
        const int c0 = (kt & 3) * 32;                 // cin base within tap
        const int tapoff = ((p / 3) * 14 + (p % 3)) * 128 + c0;  // uniform
        F16frag a;
        const bf16* ap = w2r + (size_t)(m0 + r) * 1152 + kt * 32;
        a.u[0] = *(const uint4v*)(ap + half * 8);
        a.u[1] = *(const uint4v*)(ap + 16 + half * 8);
        #pragma unroll
        for (int nt = 0; nt < 9; ++nt) {
            const bf16* bp = hbase + offs[nt] + tapoff;
            F16frag b;
            b.u[0] = *(const uint4v*)(bp);
            b.u[1] = *(const uint4v*)(bp + 16);
            acc2[nt] = wmma_bf16(a.v, b.v, acc2[nt]);
        }
    }
    #pragma unroll
    for (int nt = 0; nt < 9; ++nt) {
        #pragma unroll
        for (int i = 0; i < 8; ++i) {
            float v = acc2[nt][i] + bias2[i];
            H2[m0 + half * 8 + i][nt * 16 + r] = f2bf(v > 0.f ? v : 0.f);
        }
    }
    __syncthreads();
    // coalesced b128 writeback: H2 flat == [E][d = cout*144 + sp]
    for (int i = threadIdx.x; i < 2304; i += 256) {
        int off = i << 3;
        *(uint4v*)&outb[(size_t)e * 18432 + off] =
            *(const uint4v*)(((const bf16*)H2) + off);
    }
}

// ---------------- attention scores + softmax: one block per group -----------
__global__ __launch_bounds__(256) void attn_scores_kernel(
    const bf16* __restrict__ qb, const bf16* __restrict__ kb,
    bf16* __restrict__ abg)      // [16][128][128] attn weights out
{
    __shared__ __align__(16) char smem[128 * 1024];
    bf16  (*qc)[256] = (bf16(*)[256])(smem);            // 64 KB
    bf16  (*kc)[256] = (bf16(*)[256])(smem + 65536);    // 64 KB
    float (*sc)[128] = (float(*)[128])(smem + 65536);   // overlays kc

    const int g    = blockIdx.x;
    const int lane = threadIdx.x & 31;
    const int wid  = threadIdx.x >> 5;
    const int half = lane >> 4;
    const int r    = lane & 15;
    const int m0   = wid * 16;
    const size_t D = 18432;

    const bf16* qg = qb + (size_t)g * 128 * D;
    const bf16* kg = kb + (size_t)g * 128 * D;
    const bf16* qrow = &qc[m0 + r][0];    // kt/nt-invariant row pointers
    const bf16* krow = &kc[r][0];

    v8f acc[8] = {};
    for (int ch = 0; ch < 72; ++ch) {
        const size_t k0g = (size_t)ch * 256;
        for (int i = threadIdx.x; i < 4096; i += 256) {
            int row = i >> 5, col = (i & 31) << 3;
            size_t gi = (size_t)row * D + k0g + col;
            *(uint4v*)&qc[row][col] = *(const uint4v*)&qg[gi];
            *(uint4v*)&kc[row][col] = *(const uint4v*)&kg[gi];
        }
        if (ch + 1 < 72) {   // prefetch next chunk (global_prefetch_b8)
            int row = threadIdx.x >> 1, hh = (threadIdx.x & 1) << 7;
            size_t gp = (size_t)row * D + k0g + 256 + hh;
            __builtin_prefetch(&qg[gp], 0, 1);
            __builtin_prefetch(&kg[gp], 0, 1);
        }
        __syncthreads();
        #pragma unroll
        for (int nt = 0; nt < 8; ++nt) {
            #pragma unroll
            for (int kk = 0; kk < 8; ++kk) {
                v16bf a = load_frag(qrow,                 kk * 32, half);
                v16bf b = load_frag(krow + nt * 16 * 256, kk * 32, half);
                acc[nt] = wmma_bf16(a, b, acc[nt]);
            }
        }
        __syncthreads();
    }
    #pragma unroll
    for (int nt = 0; nt < 8; ++nt)
        #pragma unroll
        for (int i = 0; i < 8; ++i)
            sc[m0 + half * 8 + i][nt * 16 + r] = acc[nt][i];
    __syncthreads();

    // softmax rows, scale = 1/sqrt(2*128) = 1/16
    if (threadIdx.x < 128) {
        const int row = threadIdx.x;
        float mx = -3.4e38f;
        for (int j = 0; j < 128; ++j) mx = fmaxf(mx, sc[row][j]);
        float sum = 0.f;
        for (int j = 0; j < 128; ++j) {
            float e = __expf((sc[row][j] - mx) * 0.0625f);
            sc[row][j] = e;
            sum += e;
        }
        float inv = 1.f / sum;
        bf16* dst = abg + (size_t)g * 16384 + (size_t)row * 128;
        for (int j = 0; j < 128; ++j) dst[j] = f2bf(sc[row][j] * inv);
    }
}

// ---------------- attention output: grid (16 groups x 9 d-slices) -----------
__global__ __launch_bounds__(256) void attn_out_kernel(
    const bf16* __restrict__ abg, const bf16* __restrict__ vb,
    float* __restrict__ out)
{
    __shared__ __align__(16) char smem[224 * 1024];
    bf16  (*ab)[128]    = (bf16(*)[128])(smem);             // 32 KB
    bf16  (*vt)[128]    = (bf16(*)[128])(smem + 32768);     // 64 KB  v^T chunk
    float (*stage)[256] = (float(*)[256])(smem + 98304);    // 128 KB out tile

    const int g    = blockIdx.x;
    const int dc   = blockIdx.y;        // 0..8, 2048-wide d slice
    const int lane = threadIdx.x & 31;
    const int wid  = threadIdx.x >> 5;
    const int half = lane >> 4;
    const int r    = lane & 15;
    const int m0   = wid * 16;
    const size_t D = 18432;

    const bf16* vg = vb + (size_t)g * 128 * D;

    // load attention weights [128][128] bf16
    for (int i = threadIdx.x; i < 2048; i += 256) {
        int off = i << 3;
        *(uint4v*)(((bf16*)ab) + off) =
            *(const uint4v*)&abg[(size_t)g * 16384 + off];
    }
    __syncthreads();

    const bf16* arow = &ab[m0 + r][0];
    const bf16* vrow = &vt[r][0];

    for (int sub = 0; sub < 8; ++sub) {
        const size_t d0 = (size_t)dc * 2048 + sub * 256;
        // vt = v^T chunk [256 d][128 m]
        for (int i = threadIdx.x; i < 4096; i += 256) {
            int m = i >> 5, col = (i & 31) << 3;
            uint4v u = *(const uint4v*)&vg[(size_t)m * D + d0 + col];
            const bf16* h = (const bf16*)&u;
            #pragma unroll
            for (int j = 0; j < 8; ++j) vt[col + j][m] = h[j];
        }
        __syncthreads();
        for (int nt = 0; nt < 16; ++nt) {
            v8f c = {};
            #pragma unroll
            for (int kk = 0; kk < 4; ++kk) {
                v16bf a = load_frag(arow,                 kk * 32, half);
                v16bf b = load_frag(vrow + nt * 16 * 128, kk * 32, half);
                c = wmma_bf16(a, b, c);
            }
            #pragma unroll
            for (int i = 0; i < 8; ++i)
                stage[m0 + half * 8 + i][nt * 16 + r] = c[i];
        }
        __syncthreads();
        // coalesced b128 writeback of [128 rows][256 d] f32
        for (int i = threadIdx.x; i < 8192; i += 256) {
            int row = i >> 6, col = (i & 63) << 2;
            *(uint4v*)&out[((size_t)g * 128 + row) * D + d0 + col] =
                *(const uint4v*)&stage[row][col];
        }
        __syncthreads();
    }
}

extern "C" void kernel_launch(void* const* d_in, const int* in_sizes, int n_in,
                              void* d_out, int out_size, void* d_ws, size_t ws_size,
                              hipStream_t stream) {
    const float* ef = (const float*)d_in[0];
    // inputs: 2..13 = q_w1,q_b1,q_w2,q_b2, k_*, v_*
    const float* w1[3] = {(const float*)d_in[2], (const float*)d_in[6],  (const float*)d_in[10]};
    const float* b1[3] = {(const float*)d_in[3], (const float*)d_in[7],  (const float*)d_in[11]};
    const float* w2[3] = {(const float*)d_in[4], (const float*)d_in[8],  (const float*)d_in[12]};
    const float* b2[3] = {(const float*)d_in[5], (const float*)d_in[9],  (const float*)d_in[13]};

    char* ws = (char*)d_ws;
    const size_t w1rBytes = (size_t)128 * 160 * 2;          // per qkv
    const size_t w2rBytes = (size_t)128 * 1152 * 2;         // per qkv
    const size_t bufElems = (size_t)2048 * 18432;
    bf16* w1r = (bf16*)ws;
    bf16* w2r = (bf16*)(ws + 3 * w1rBytes);
    bf16* qkv = (bf16*)(ws + 3 * w1rBytes + 3 * w2rBytes);
    bf16* abg = (bf16*)(ws + 3 * w1rBytes + 3 * w2rBytes + 3 * bufElems * 2);

    for (int j = 0; j < 3; ++j) {
        pack_w1_kernel<<<80, 256, 0, stream>>>(w1[j], w1r + (size_t)j * 128 * 160);
        pack_w2_kernel<<<576, 256, 0, stream>>>(w2[j], w2r + (size_t)j * 128 * 1152);
    }
    for (int j = 0; j < 3; ++j) {
        conv_block_kernel<<<2048, 256, 0, stream>>>(
            ef, w1r + (size_t)j * 128 * 160, b1[j],
            w2r + (size_t)j * 128 * 1152, b2[j], qkv + (size_t)j * bufElems);
    }
    attn_scores_kernel<<<16, 256, 0, stream>>>(qkv, qkv + bufElems, abg);
    attn_out_kernel<<<dim3(16, 9), 256, 0, stream>>>(abg, qkv + 2 * bufElems,
                                                     (float*)d_out);
}
